// GenericMXMNet_NoLocalInfo_16947940950126
// MI455X (gfx1250) — matrix operand order repaired and checked by reading, not compiled, not run on hardware
//
#include <hip/hip_runtime.h>

// ---------------- problem constants (from reference) ----------------
#define DIMC   128
#define NLAY   6
#define HEADS  8
#define NPER   32
#define NGRAPH 512
#define NNODE  (NGRAPH * NPER)          // 16384
#define EPERG  (NPER * (NPER - 1))      // 992
#define NEDGE  (NGRAPH * EPERG)        // 507904
#define CCH    (DIMC * NLAY)            // 768

typedef __bf16 bf16_t;
typedef __attribute__((ext_vector_type(16))) __bf16 v16bf;
typedef __attribute__((ext_vector_type(8)))  float  v8f;

#if __has_builtin(__builtin_amdgcn_global_load_async_to_lds_b128) && \
    __has_builtin(__builtin_amdgcn_s_wait_asynccnt)
#define HAVE_ASYNC 1
#else
#define HAVE_ASYNC 0
#endif

// b128 async payload type per the builtin's prototype: int vector_size(16),
// src in AS1 (__device__/global), dst in AS3 (LDS).
typedef int v4i_pay __attribute__((vector_size(16)));
typedef __attribute__((address_space(1))) v4i_pay as1_v4i;
typedef __attribute__((address_space(3))) v4i_pay as3_v4i;
#define AS1P(p) ((as1_v4i*)(unsigned long long)(const void*)(p))
#define AS3P(p) ((as3_v4i*)(unsigned long long)(const void*)(p))

__device__ __forceinline__ float wave_sum(float v) {
  #pragma unroll
  for (int m = 16; m > 0; m >>= 1) v += __shfl_xor(v, m, 32);
  return v;
}
__device__ __forceinline__ float wave_max(float v) {
  #pragma unroll
  for (int m = 16; m > 0; m >>= 1) v = fmaxf(v, __shfl_xor(v, m, 32));
  return v;
}
__device__ __forceinline__ float silu_f(float x) { return x / (1.0f + __expf(-x)); }

// ======================================================================
// WMMA GEMM:  C[M,N] = act( A[M,K](f32->bf16) @ B[K,N](f32->bf16) + bias
//                           + res[row*resLd + n] )
// Block tile 64x128, 128 threads = 4 waves, each wave 32x64 via 2x4
// v_wmma_f32_16x16x32_bf16.  fp32 tiles live in LDS (fed by CDNA5
// GLOBAL_LOAD_ASYNC_TO_LDS_B128 double-buffered DMA when available);
// bf16 conversion happens at fragment build (v_cvt_pk_bf16_f32).
// Requires M%64==0, N%128==0, K%32==0 (true for every call site here).
// ======================================================================
#define GT_M 64
#define GT_N 128
#define GT_K 32
#define LDA_S 36    // 32 + 4 pad (f32 elems, keeps 16B alignment)
#define LDB_S 132   // 128 + 4 pad
#if HAVE_ASYNC
#define BUFS 2
#else
#define BUFS 1
#endif

__global__ __launch_bounds__(128)
void k_gemm(const float* __restrict__ A, int lda,
            const float* __restrict__ B, int ldb,
            const float* __restrict__ bias,
            const float* __restrict__ res, int resLd,
            float* __restrict__ C, int ldc,
            int M, int N, int K, int act)
{
  __shared__ float As[BUFS][GT_M * LDA_S];
  __shared__ float Bs[BUFS][GT_K * LDB_S];

  const int tid  = threadIdx.x;
  const int lane = tid & 31;
  const int wv   = tid >> 5;              // 0..3
  const int bm   = blockIdx.x * GT_M;
  const int bn   = blockIdx.y * GT_N;
  const int waveM = (wv >> 1) * 32;
  const int waveN = (wv & 1) * 64;
  const int lm   = lane & 15;
  const int lhi  = lane >> 4;             // 0/1

  v8f acc[2][4];
  #pragma unroll
  for (int i = 0; i < 2; i++)
    #pragma unroll
    for (int j = 0; j < 4; j++) acc[i][j] = v8f{};

  // fragment build + 2x4 WMMA from fp32 LDS tiles
  auto compute = [&](int buf) {
    v16bf afr[2], bfr[4];
    #pragma unroll
    for (int mt = 0; mt < 2; mt++) {
      const float* ar = &As[buf][(waveM + mt * 16 + lm) * LDA_S + 8 * lhi];
      #pragma unroll
      for (int e = 0; e < 8; e++) {
        afr[mt][e]     = (bf16_t)ar[e];
        afr[mt][e + 8] = (bf16_t)ar[e + 16];
      }
    }
    #pragma unroll
    for (int nt = 0; nt < 4; nt++) {
      const float* br = &Bs[buf][(8 * lhi) * LDB_S + waveN + nt * 16 + lm];
      #pragma unroll
      for (int e = 0; e < 8; e++) {
        bfr[nt][e]     = (bf16_t)br[e * LDB_S];
        bfr[nt][e + 8] = (bf16_t)br[(e + 16) * LDB_S];
      }
    }
    #pragma unroll
    for (int mt = 0; mt < 2; mt++)
      #pragma unroll
      for (int nt = 0; nt < 4; nt++)
        acc[mt][nt] = __builtin_amdgcn_wmma_f32_16x16x32_bf16(
            false, afr[mt], false, bfr[nt], (short)0, acc[mt][nt], false, false);
  };

#if HAVE_ASYNC
  // ---- CDNA5 async DMA tile fill: 12 x b128 per wave per K-step ----
  auto issue_tile = [&](int k0, int buf) {
    #pragma unroll
    for (int i = 0; i < 4; i++) {                 // A: 64x32 f32 = 512 chunks
      int c = i * 128 + tid;
      int row = c >> 3, c4 = (c & 7) * 4;
      const float* gp = A + (size_t)(bm + row) * lda + k0 + c4;
      float* lp = &As[buf][row * LDA_S + c4];
      __builtin_amdgcn_global_load_async_to_lds_b128(AS1P(gp), AS3P(lp), 0, 0);
    }
    #pragma unroll
    for (int i = 0; i < 8; i++) {                 // B: 32x128 f32 = 1024 chunks
      int c = i * 128 + tid;
      int row = c >> 5, c4 = (c & 31) * 4;
      const float* gp = B + (size_t)(k0 + row) * ldb + bn + c4;
      float* lp = &Bs[buf][row * LDB_S + c4];
      __builtin_amdgcn_global_load_async_to_lds_b128(AS1P(gp), AS3P(lp), 0, 0);
    }
  };

  issue_tile(0, 0);
  int cur = 0;
  for (int k0 = 0; k0 < K; k0 += GT_K) {
    if (k0 + GT_K < K) {
      issue_tile(k0 + GT_K, cur ^ 1);             // prefetch next tile
      __builtin_amdgcn_s_wait_asynccnt(12);       // retire current tile's 12 ops
    } else {
      __builtin_amdgcn_s_wait_asynccnt(0);
    }
    __syncthreads();                              // DMA->LDS visible to all waves
    compute(cur);
    __syncthreads();                              // reads done before buffer reuse
    cur ^= 1;
  }
#else
  // ---- fallback: synchronous fill + L2 prefetch of the next tile ----
  for (int k0 = 0; k0 < K; k0 += GT_K) {
    #pragma unroll
    for (int i = 0; i < 4; i++) {
      int c = i * 128 + tid;
      int row = c >> 3, c4 = (c & 7) * 4;
      const float* gp = A + (size_t)(bm + row) * lda + k0 + c4;
      *(float4*)&As[0][row * LDA_S + c4] = *(const float4*)gp;
      if (k0 + GT_K < K) __builtin_prefetch(gp + GT_K, 0, 1);
    }
    #pragma unroll
    for (int i = 0; i < 8; i++) {
      int c = i * 128 + tid;
      int row = c >> 5, c4 = (c & 31) * 4;
      const float* gp = B + (size_t)(k0 + row) * ldb + bn + c4;
      *(float4*)&Bs[0][row * LDB_S + c4] = *(const float4*)gp;
      if (k0 + GT_K < K) __builtin_prefetch(gp + (size_t)GT_K * ldb, 0, 1);
    }
    __syncthreads();
    compute(0);
    __syncthreads();
  }
#endif

  // ---- epilogue: C/D layout lane<16: m=r,n=lane ; lane>=16: m=r+8,n=lane-16
  #pragma unroll
  for (int mt = 0; mt < 2; mt++) {
    #pragma unroll
    for (int nt = 0; nt < 4; nt++) {
      #pragma unroll
      for (int r = 0; r < 8; r++) {
        int m = bm + waveM + mt * 16 + r + 8 * lhi;
        int n = bn + waveN + nt * 16 + lm;
        float x = acc[mt][nt][r];
        if (bias) x += bias[n];
        if (res)  x += res[(size_t)m * resLd + n];
        if (act == 1) x = silu_f(x);
        else if (act == 2) x = fmaxf(x, 0.0f);
        C[(size_t)m * ldc + n] = x;
      }
    }
  }
}

// ======================================================================
// preprocessing kernels
// ======================================================================
__global__ __launch_bounds__(256)
void k_embed(const int* __restrict__ x, const float* __restrict__ emb,
             float* __restrict__ h0)
{
  int idx = blockIdx.x * 256 + threadIdx.x;          // N*128
  if (idx >= NNODE * DIMC) return;
  int i = idx >> 7, c = idx & 127;
  h0[idx] = emb[x[i] * DIMC + c];
}

// edge RBF raw features + batchnorm sum/sumsq accumulation
__global__ __launch_bounds__(256)
void k_rbf_stats(const float* __restrict__ pos, const int* __restrict__ src,
                 const int* __restrict__ dst, const float* __restrict__ freq,
                 float* __restrict__ rbf_raw, float* __restrict__ stats)
{
  __shared__ float ssum[16], ssq[16];
  int tid = threadIdx.x;
  if (tid < 16) { ssum[tid] = 0.0f; ssq[tid] = 0.0f; }
  __syncthreads();

  int e = blockIdx.x * 256 + tid;
  float loc[16];
  if (e < NEDGE) {
    int sN = src[e], dN = dst[e];
    float dx = pos[dN * 3 + 0] - pos[sN * 3 + 0];
    float dy = pos[dN * 3 + 1] - pos[sN * 3 + 1];
    float dz = pos[dN * 3 + 2] - pos[sN * 3 + 2];
    float d  = sqrtf(dx * dx + dy * dy + dz * dz) + 1e-8f;
    float dn = d * 0.2f;                               // / CUT(5)
    float dn2 = dn * dn, dn4 = dn2 * dn2, dn5 = dn4 * dn;
    float env = 1.0f / dn - 28.0f * dn5 + 48.0f * dn5 * dn - 21.0f * dn5 * dn2;
    #pragma unroll
    for (int j = 0; j < 16; j++) {
      loc[j] = env * __sinf(freq[j] * dn);
      rbf_raw[(size_t)e * 16 + j] = loc[j];
    }
  } else {
    #pragma unroll
    for (int j = 0; j < 16; j++) loc[j] = 0.0f;
  }
  int lane = tid & 31;
  #pragma unroll
  for (int j = 0; j < 16; j++) {
    float v  = wave_sum(loc[j]);
    float v2 = wave_sum(loc[j] * loc[j]);
    if (lane == 0) { atomicAdd(&ssum[j], v); atomicAdd(&ssq[j], v2); }
  }
  __syncthreads();
  if (tid < 16) { atomicAdd(&stats[tid], ssum[tid]); atomicAdd(&stats[16 + tid], ssq[tid]); }
}

// normalize + rbf@rbf_W + rbf_b, SiLU  (K=16: VALU, memory bound)
__global__ __launch_bounds__(128)
void k_rbf_feat(const float* __restrict__ rbf_raw, const float* __restrict__ stats,
                const float* __restrict__ bn_g, const float* __restrict__ bn_b,
                const float* __restrict__ W, const float* __restrict__ bvec,
                float* __restrict__ efeat)
{
  __shared__ float sh[16];
  int e = blockIdx.x;
  int t = threadIdx.x;
  if (t < 16) {
    float mu  = stats[t] * (1.0f / (float)NEDGE);
    float var = stats[16 + t] * (1.0f / (float)NEDGE) - mu * mu;
    sh[t] = (rbf_raw[(size_t)e * 16 + t] - mu) * rsqrtf(var + 1e-5f) * bn_g[t] + bn_b[t];
  }
  __syncthreads();
  float acc = bvec[t];
  #pragma unroll
  for (int j = 0; j < 16; j++) acc += sh[j] * W[j * DIMC + t];
  efeat[(size_t)e * DIMC + t] = silu_f(acc);
}

// ======================================================================
// message-passing attention
// ======================================================================
// per-(edge,head) score: s = <q[dst,h,:], kx[src,h,:] + ek[e,h,:]> / 4
__global__ __launch_bounds__(256)
void k_edge_score(const float* __restrict__ q, const float* __restrict__ kx,
                  const float* __restrict__ ek, const int* __restrict__ src,
                  const int* __restrict__ dst, float* __restrict__ s_edge)
{
  int idx = blockIdx.x * 256 + threadIdx.x;
  if (idx >= NEDGE * HEADS) return;
  int e = idx >> 3, hh = idx & 7;
  int sN = src[e], dN = dst[e];
  const float* qp = q  + (size_t)dN * DIMC + hh * 16;
  const float* kp = kx + (size_t)sN * DIMC + hh * 16;
  const float* ep = ek + (size_t)e  * DIMC + hh * 16;
  float s = 0.0f;
  #pragma unroll
  for (int d = 0; d < 16; d++) s += qp[d] * (kp[d] + ep[d]);
  s_edge[idx] = s * 0.25f;                     // 1/sqrt(16)
}

// per-(node,head) wave: segment softmax over 31 in-edges + weighted v agg
__global__ __launch_bounds__(256)
void k_attn_agg(const float* __restrict__ s_edge, const float* __restrict__ vx,
                const float* __restrict__ ev, float* __restrict__ agg)
{
  __shared__ float sal[8][32];
  __shared__ int   sei[8][32];
  int wv = threadIdx.x >> 5, lane = threadIdx.x & 31;
  int wid = blockIdx.x * 8 + wv;               // = i*HEADS + hh
  int hh = wid & 7;
  int i  = wid >> 3;
  int g  = i >> 5, li = i & 31;
  int a  = lane;                                // source-node index in graph
  bool valid = (a != li);
  int eidx = g * EPERG + a * (NPER - 1) + li - (li > a ? 1 : 0);
  if (!valid) eidx = g * EPERG;
  float s  = valid ? s_edge[(size_t)eidx * HEADS + hh] : -1e30f;
  float m  = wave_max(s);
  float ex = valid ? __expf(s - m) : 0.0f;
  float den = wave_sum(ex);
  sal[wv][lane] = ex / (den + 1e-16f);
  sei[wv][lane] = eidx;
  __syncthreads();
  if (lane < 16) {
    int d = hh * 16 + lane;
    float acc = 0.0f;
    #pragma unroll 4
    for (int a2 = 0; a2 < 32; a2++) {
      if (a2 == li) continue;
      int e2 = sei[wv][a2];
      int s2 = g * NPER + a2;
      acc += sal[wv][a2] * (vx[(size_t)s2 * DIMC + d] + ev[(size_t)e2 * DIMC + d]);
    }
    agg[(size_t)i * DIMC + d] = acc;
  }
}

// ======================================================================
// readout kernels
// ======================================================================
// LayerNorm over C=768, wave per row; optional elementwise residual
__global__ __launch_bounds__(256)
void k_ln(const float* __restrict__ A, const float* __restrict__ R,
          const float* __restrict__ g, const float* __restrict__ b,
          float* __restrict__ O, int rows)
{
  int wid = blockIdx.x * 8 + (threadIdx.x >> 5);
  if (wid >= rows) return;
  int lane = threadIdx.x & 31;
  const float* a = A + (size_t)wid * CCH;
  const float* r = R ? R + (size_t)wid * CCH : nullptr;
  float x[24];
  float s = 0.0f, s2 = 0.0f;
  #pragma unroll
  for (int t = 0; t < 24; t++) {
    float v = a[lane + 32 * t];
    if (r) v += r[lane + 32 * t];
    x[t] = v; s += v; s2 += v * v;
  }
  s = wave_sum(s); s2 = wave_sum(s2);
  float mu  = s * (1.0f / (float)CCH);
  float var = s2 * (1.0f / (float)CCH) - mu * mu;
  float inv = rsqrtf(var + 1e-5f);
  float* o = O + (size_t)wid * CCH;
  #pragma unroll
  for (int t = 0; t < 24; t++) {
    int c = lane + 32 * t;
    o[c] = (x[t] - mu) * inv * g[c] + b[c];
  }
}

// SAB attention: wave per (graph, query, head); 32 keys on lanes, Dh=96
__global__ __launch_bounds__(256)
void k_attn_enc(const float* __restrict__ q, const float* __restrict__ k,
                const float* __restrict__ v, float* __restrict__ o)
{
  __shared__ float sal[8][32];
  int wv = threadIdx.x >> 5, lane = threadIdx.x & 31;
  int wid = blockIdx.x * 8 + wv;               // qrow*HEADS + hh
  int hh = wid & 7;
  int qrow = wid >> 3;
  int g = qrow >> 5;
  const float* qp = q + (size_t)qrow * CCH + hh * 96;
  const float* kp = k + (size_t)(g * NPER + lane) * CCH + hh * 96;
  float s = 0.0f;
  #pragma unroll 8
  for (int d = 0; d < 96; d++) s += qp[d] * kp[d];
  s *= 0.1020620726159658f;                    // 1/sqrt(96)
  float m = wave_max(s);
  float ex = __expf(s - m);
  float den = wave_sum(ex);
  sal[wv][lane] = ex / den;
  __syncthreads();
  float o0 = 0.0f, o1 = 0.0f, o2 = 0.0f;
  for (int kj = 0; kj < 32; kj++) {
    float al = sal[wv][kj];
    const float* vp = v + (size_t)(g * NPER + kj) * CCH + hh * 96 + lane * 3;
    o0 += al * vp[0]; o1 += al * vp[1]; o2 += al * vp[2];
  }
  float* op = o + (size_t)qrow * CCH + hh * 96 + lane * 3;
  op[0] = o0; op[1] = o1; op[2] = o2;
}

// PMA attention: wave per (graph, head); single broadcast-seed query
__global__ __launch_bounds__(256)
void k_attn_pma(const float* __restrict__ qseed, const float* __restrict__ k,
                const float* __restrict__ v, float* __restrict__ o)
{
  __shared__ float sal[8][32];
  int wv = threadIdx.x >> 5, lane = threadIdx.x & 31;
  int wid = blockIdx.x * 8 + wv;               // g*HEADS + hh
  int hh = wid & 7;
  int g = wid >> 3;
  const float* qp = qseed + hh * 96;
  const float* kp = k + (size_t)(g * NPER + lane) * CCH + hh * 96;
  float s = 0.0f;
  #pragma unroll 8
  for (int d = 0; d < 96; d++) s += qp[d] * kp[d];
  s *= 0.1020620726159658f;
  float m = wave_max(s);
  float ex = __expf(s - m);
  float den = wave_sum(ex);
  sal[wv][lane] = ex / den;
  __syncthreads();
  float o0 = 0.0f, o1 = 0.0f, o2 = 0.0f;
  for (int kj = 0; kj < 32; kj++) {
    float al = sal[wv][kj];
    const float* vp = v + (size_t)(g * NPER + kj) * CCH + hh * 96 + lane * 3;
    o0 += al * vp[0]; o1 += al * vp[1]; o2 += al * vp[2];
  }
  float* op = o + (size_t)g * CCH + hh * 96 + lane * 3;
  op[0] = o0; op[1] = o1; op[2] = o2;
}

// qseed[n] = seed @ Wq + bq   (M=1 GEMV)
__global__ __launch_bounds__(256)
void k_seed_gemv(const float* __restrict__ seed, const float* __restrict__ Wq,
                 const float* __restrict__ bq, float* __restrict__ qseed)
{
  int n = blockIdx.x * 256 + threadIdx.x;
  if (n >= CCH) return;
  float acc = bq[n];
  for (int c = 0; c < CCH; c++) acc += seed[c] * Wq[c * CCH + n];
  qseed[n] = acc;
}

// ======================================================================
// host orchestration
// ======================================================================
struct MabP {
  const float *Wk, *Wl, *Wo, *Wq, *Wv;
  const float *be1, *be2, *bk, *bl, *bo, *bq, *bv, *g1, *g2;
};

extern "C" void kernel_launch(void* const* d_in, const int* in_sizes, int n_in,
                              void* d_out, int out_size, void* d_ws, size_t ws_size,
                              hipStream_t stream)
{
  (void)in_sizes; (void)n_in; (void)out_size; (void)ws_size;

  // ------- input unpack (top-level dict insertion order; params pytree
  //         flattened jax-style: nested dict keys sorted alphabetically) ----
  const int*   x    = (const int*)d_in[0];
  const float* pos  = (const float*)d_in[1];
  /* batch = d_in[2] unused */
  const int*   src  = (const int*)d_in[3];
  const int*   dst  = (const int*)d_in[4];
  const int P = 5;
  auto F = [&](int i) { return (const float*)d_in[P + i]; };
  auto mab = [&](int base) {
    MabP m;
    m.Wk = F(base + 0);  m.Wl = F(base + 1);  m.Wo = F(base + 2);
    m.Wq = F(base + 3);  m.Wv = F(base + 4);
    m.be1 = F(base + 5); m.be2 = F(base + 6); m.bk = F(base + 7);
    m.bl = F(base + 8);  m.bo = F(base + 9);  m.bq = F(base + 10);
    m.bv = F(base + 11); m.g1 = F(base + 12); m.g2 = F(base + 13);
    return m;
  };
  const float* bn_b = F(0);
  const float* bn_g = F(1);
  MabP dec  = mab(2);
  const float* emb  = F(16);
  MabP enc  = mab(17);
  const float* freq = F(31);
  // layers: sorted keys [W1,W2,Wek,Wev,Wk,Wo,Wq,Wv,b1,b2], 10 each
  const float* LW1[NLAY]; const float* LW2[NLAY]; const float* LWek[NLAY];
  const float* LWev[NLAY]; const float* LWk[NLAY]; const float* LWo[NLAY];
  const float* LWq[NLAY]; const float* LWv[NLAY]; const float* Lb1[NLAY];
  const float* Lb2[NLAY];
  for (int l = 0; l < NLAY; l++) {
    int b = 32 + 10 * l;
    LW1[l] = F(b + 0); LW2[l] = F(b + 1); LWek[l] = F(b + 2); LWev[l] = F(b + 3);
    LWk[l] = F(b + 4); LWo[l] = F(b + 5); LWq[l]  = F(b + 6); LWv[l]  = F(b + 7);
    Lb1[l] = F(b + 8); Lb2[l] = F(b + 9);
  }
  const float* out_W = F(92);
  const float* out_b = F(93);
  MabP pma   = mab(94);
  const float* pma_lin_W = F(108);
  const float* pma_lin_b = F(109);
  const float* rbf_W = F(110);
  const float* rbf_b = F(111);
  const float* seed  = F(112);

  // ------------------- workspace carving (floats) -------------------
  float* W = (float*)d_ws;
  size_t off = 0;
  auto carve = [&](size_t n) { size_t o = off; off += (n + 63) & ~(size_t)63; return W + o; };
  float* stats = carve(32);
  float* h0    = carve((size_t)NNODE * DIMC);
  float* hcat  = carve((size_t)NNODE * CCH);
  float* nq    = carve((size_t)NNODE * DIMC);
  float* nk    = carve((size_t)NNODE * DIMC);
  float* nv    = carve((size_t)NNODE * DIMC);
  float* nagg  = carve((size_t)NNODE * DIMC);
  float* nmid  = carve((size_t)NNODE * DIMC);
  float* nt    = carve((size_t)NNODE * DIMC);
  float* sedge = carve((size_t)NEDGE * HEADS);
  float* efeat = carve((size_t)NEDGE * DIMC);   // aliased by r0..r3 in readout
  float* ekv   = carve((size_t)NEDGE * DIMC);   // ek then ev; aliases rbf_raw & pma pool

  float* rbf_raw = ekv;                                   // [E,16], pre-layer only
  // readout 768-wide buffers alias edge-feature region (dead after layers)
  float* r0 = efeat;
  float* r1 = efeat + (size_t)NNODE * CCH;
  float* r2 = efeat + 2 * (size_t)NNODE * CCH;
  float* r3 = efeat + 3 * (size_t)NNODE * CCH;
  // small PMA/dec pool aliases ekv region
  float* qseed = ekv;
  float* t0 = ekv + 1024;
  float* t1 = t0 + (size_t)NGRAPH * CCH;
  float* t2 = t1 + (size_t)NGRAPH * CCH;
  float* t3 = t2 + (size_t)NGRAPH * CCH;
  float* t4 = t3 + (size_t)NGRAPH * CCH;

  auto gemm = [&](const float* A, int lda, const float* B, int ldb,
                  const float* bias, const float* res, int resLd,
                  float* Cc, int ldc, int M, int N, int K, int act) {
    dim3 grid((M + GT_M - 1) / GT_M, (N + GT_N - 1) / GT_N);
    k_gemm<<<grid, dim3(128), 0, stream>>>(A, lda, B, ldb, bias, res, resLd,
                                           Cc, ldc, M, N, K, act);
  };

  // ---------------------- preprocessing ----------------------
  (void)hipMemsetAsync(stats, 0, 32 * sizeof(float), stream);
  k_embed<<<(NNODE * DIMC + 255) / 256, 256, 0, stream>>>(x, emb, h0);
  k_rbf_stats<<<(NEDGE + 255) / 256, 256, 0, stream>>>(pos, src, dst, freq,
                                                       rbf_raw, stats);
  k_rbf_feat<<<NEDGE, 128, 0, stream>>>(rbf_raw, stats, bn_g, bn_b,
                                        rbf_W, rbf_b, efeat);

  // ---------------------- message-passing layers ----------------------
  for (int l = 0; l < NLAY; l++) {
    const float* hin = (l == 0) ? h0 : (hcat + (size_t)(l - 1) * DIMC);
    int ldh = (l == 0) ? DIMC : CCH;
    gemm(hin, ldh, LWq[l], DIMC, nullptr, nullptr, 0, nq, DIMC, NNODE, DIMC, DIMC, 0);
    gemm(hin, ldh, LWk[l], DIMC, nullptr, nullptr, 0, nk, DIMC, NNODE, DIMC, DIMC, 0);
    gemm(hin, ldh, LWv[l], DIMC, nullptr, nullptr, 0, nv, DIMC, NNODE, DIMC, DIMC, 0);
    // ek = efeat @ Wek  (big [E,128]x[128,128] WMMA GEMM)
    gemm(efeat, DIMC, LWek[l], DIMC, nullptr, nullptr, 0, ekv, DIMC, NEDGE, DIMC, DIMC, 0);
    k_edge_score<<<(NEDGE * HEADS + 255) / 256, 256, 0, stream>>>(nq, nk, ekv,
                                                                 src, dst, sedge);
    // ev = efeat @ Wev  (reuses ekv buffer; lifetimes disjoint)
    gemm(efeat, DIMC, LWev[l], DIMC, nullptr, nullptr, 0, ekv, DIMC, NEDGE, DIMC, DIMC, 0);
    k_attn_agg<<<NNODE, 256, 0, stream>>>(sedge, nv, ekv, nagg);
    // h = h + agg @ Wo
    gemm(nagg, DIMC, LWo[l], DIMC, nullptr, hin, ldh, nmid, DIMC, NNODE, DIMC, DIMC, 0);
    // h = h + silu(h@W1+b1)@W2 + b2  -> written into hcat column block l
    gemm(nmid, DIMC, LW1[l], DIMC, Lb1[l], nullptr, 0, nt, DIMC, NNODE, DIMC, DIMC, 1);
    gemm(nt, DIMC, LW2[l], DIMC, Lb2[l], nmid, DIMC,
         hcat + (size_t)l * DIMC, CCH, NNODE, DIMC, DIMC, 0);
  }

  // ---------------------- SAB encoder (MAB(mol, mol)) ----------------------
  gemm(hcat, CCH, enc.Wq, CCH, enc.bq, nullptr, 0, r0, CCH, NNODE, CCH, CCH, 0);
  gemm(hcat, CCH, enc.Wk, CCH, enc.bk, nullptr, 0, r1, CCH, NNODE, CCH, CCH, 0);
  gemm(hcat, CCH, enc.Wv, CCH, enc.bv, nullptr, 0, r2, CCH, NNODE, CCH, CCH, 0);
  k_attn_enc<<<NNODE * HEADS / 8, 256, 0, stream>>>(r0, r1, r2, r3);
  gemm(r3, CCH, enc.Wo, CCH, enc.bo, hcat, CCH, r0, CCH, NNODE, CCH, CCH, 0);
  k_ln<<<(NNODE + 7) / 8, 256, 0, stream>>>(r0, nullptr, enc.g1, enc.be1, r1, NNODE);
  gemm(r1, CCH, enc.Wl, CCH, enc.bl, nullptr, 0, r2, CCH, NNODE, CCH, CCH, 2);
  k_ln<<<(NNODE + 7) / 8, 256, 0, stream>>>(r1, r2, enc.g2, enc.be2, r0, NNODE);
  // henc = r0

  // ---------------------- PMA ----------------------
  gemm(r0, CCH, pma_lin_W, CCH, pma_lin_b, nullptr, 0, r2, CCH, NNODE, CCH, CCH, 2); // y
  gemm(r2, CCH, pma.Wk, CCH, pma.bk, nullptr, 0, r3, CCH, NNODE, CCH, CCH, 0);
  gemm(r2, CCH, pma.Wv, CCH, pma.bv, nullptr, 0, r1, CCH, NNODE, CCH, CCH, 0);
  k_seed_gemv<<<(CCH + 255) / 256, 256, 0, stream>>>(seed, pma.Wq, pma.bq, qseed);
  k_attn_pma<<<NGRAPH * HEADS / 8, 256, 0, stream>>>(qseed, r3, r1, t0);
  gemm(t0, CCH, pma.Wo, CCH, pma.bo, seed, 0 /*broadcast row*/, t1, CCH, NGRAPH, CCH, CCH, 0);
  k_ln<<<NGRAPH / 8, 256, 0, stream>>>(t1, nullptr, pma.g1, pma.be1, t2, NGRAPH);
  gemm(t2, CCH, pma.Wl, CCH, pma.bl, nullptr, 0, t3, CCH, NGRAPH, CCH, CCH, 2);
  k_ln<<<NGRAPH / 8, 256, 0, stream>>>(t2, t3, pma.g2, pma.be2, t4, NGRAPH);
  // hp = t4

  // ------------- decoder MAB(hp,hp): Sq=Sk=1 => softmax==1 => o = v -------------
  gemm(t4, CCH, dec.Wv, CCH, dec.bv, nullptr, 0, t0, CCH, NGRAPH, CCH, CCH, 0);
  gemm(t0, CCH, dec.Wo, CCH, dec.bo, t4, CCH, t1, CCH, NGRAPH, CCH, CCH, 0);
  k_ln<<<NGRAPH / 8, 256, 0, stream>>>(t1, nullptr, dec.g1, dec.be1, t2, NGRAPH);
  gemm(t2, CCH, dec.Wl, CCH, dec.bl, nullptr, 0, t3, CCH, NGRAPH, CCH, CCH, 2);
  k_ln<<<NGRAPH / 8, 256, 0, stream>>>(t2, t3, dec.g2, dec.be2, t0, NGRAPH);

  // ---------------------- output head ----------------------
  gemm(t0, CCH, out_W, DIMC, out_b, nullptr, 0, (float*)d_out, DIMC,
       NGRAPH, DIMC, CCH, 0);
}
